// GIBModel_12378095747458
// MI455X (gfx1250) — matrix operand order repaired and checked by reading, not compiled, not run on hardware
//
#include <hip/hip_runtime.h>
#include <math.h>

// ---------------- problem constants ----------------
static constexpr int N_NODES = 30000;
static constexpr int MP      = 30080;          // M padded to multiple of 128
static constexpr int NB      = 150;
static constexpr int SZ      = 200;
static constexpr int NE      = 240000;
static constexpr int F_IN    = 768;
static constexpr int HID     = 256;
static constexpr int HEADS   = 4;
static constexpr int KCH     = 10;
static constexpr int ET      = NE + N_NODES;   // edges + self loops

// ---------------- CDNA5 WMMA types ----------------
typedef __attribute__((ext_vector_type(16))) __bf16        v16bf;
typedef __attribute__((ext_vector_type(8)))  float         v8f;
typedef __attribute__((ext_vector_type(4)))  unsigned int  v4u;
typedef __attribute__((ext_vector_type(4)))  unsigned int  u32x4;
typedef __attribute__((ext_vector_type(8)))  unsigned int  u32x8;

union FragBF { v4u u[2]; v16bf bf; };

__device__ __forceinline__ unsigned short f2bf(float f) {
  unsigned int u = __float_as_uint(f);
  unsigned int r = u + 0x7FFFu + ((u >> 16) & 1u);   // round-to-nearest-even
  return (unsigned short)(r >> 16);
}

__device__ __forceinline__ void atomicMaxF(float* addr, float v) {
  if (v >= 0.f) atomicMax((int*)addr, __float_as_int(v));
  else          atomicMin((unsigned int*)addr, __float_as_uint(v));
}

// ---------------- Tensor Data Mover: 2D tile Global -> LDS ----------------
// D# per CDNA5 ISA ch.8: group0 = {count, lds_addr, global_addr, type=2},
// group1 = {data_size=2B, tensor_dim0/1, tile_dim0/1, tensor_dim0_stride}.
// tile_dim0 elements are contiguous in memory and packed row-major in LDS.
__device__ __forceinline__ void tdm_load_2d(unsigned lds_byte_addr, const void* gaddr,
                                            unsigned tile_d0, unsigned tile_d1,
                                            unsigned stride_elems,
                                            unsigned tensor_d0, unsigned tensor_d1) {
  unsigned long long ga = (unsigned long long)(size_t)gaddr;
  u32x4 g0;
  g0[0] = 1u;                                             // count=1, user mode
  g0[1] = lds_byte_addr;                                  // LDS byte address
  g0[2] = (unsigned)ga;                                   // global addr [31:0]
  g0[3] = ((unsigned)(ga >> 32) & 0x01FFFFFFu) | (2u << 30); // [56:32] | type=2
  u32x8 g1;
  g1[0] = 1u << 16;                                       // data_size=1 -> 2 bytes
  g1[1] = (tensor_d0 & 0xFFFFu) << 16;                    // tensor_dim0[15:0]
  g1[2] = (tensor_d0 >> 16) | ((tensor_d1 & 0xFFFFu) << 16);
  g1[3] = (tensor_d1 >> 16) | ((tile_d0 & 0xFFFFu) << 16);
  g1[4] = tile_d1 & 0xFFFFu;                              // tile_dim1, tile_dim2=0
  g1[5] = stride_elems;                                   // dim0 stride [31:0]
  g1[6] = 0;                                              // stride hi, dim1 stride
  g1[7] = 0;
  asm volatile("tensor_load_to_lds %0, %1" :: "s"(g0), "s"(g1) : "memory");
}

// ---------------- utility kernels ----------------
__global__ void fill_f32(float* __restrict__ p, float v, size_t n) {
  size_t i = (size_t)blockIdx.x * blockDim.x + threadIdx.x;
  if (i < n) p[i] = v;
}

__global__ void cvt_bf16(const float* __restrict__ src, unsigned short* __restrict__ dst, size_t n) {
  size_t i = (size_t)blockIdx.x * blockDim.x + threadIdx.x;
  if (i < n) dst[i] = f2bf(src[i]);
}

// W is K x N row-major f32; dst is N x K row-major bf16 (B transposed)
__global__ void cvt_bf16_transpose(const float* __restrict__ src, unsigned short* __restrict__ dst,
                                   int K, int N) {
  size_t i = (size_t)blockIdx.x * blockDim.x + threadIdx.x;
  size_t total = (size_t)K * N;
  if (i >= total) return;
  int k = (int)(i / N), n = (int)(i % N);
  dst[(size_t)n * K + k] = f2bf(src[i]);
}

// ---------------- bf16 WMMA GEMM, double-buffered TDM pipeline ----------------
// C[M x N] = A[M x K] * B[K x N]; A row-major bf16, Bt = B^T (N x K) bf16.
// Block: 8 waves -> 128(M) x 64(N) tile; wave (wr,wc) owns a 32x32 patch
// (2x2 WMMA accumulators). K staged in 64-wide chunks through LDS via TDM,
// with the DMA for chunk t+1 overlapped with the WMMA work on chunk t.
static constexpr int BM = 128, BN = 64, BK = 64;

__global__ __launch_bounds__(256) void gemm_bf16_wmma(
    const unsigned short* __restrict__ A, const unsigned short* __restrict__ Bt,
    float* __restrict__ C, int M, int K, int N) {
  __shared__ unsigned short ldsA[2][BM * BK];   // 2 x 16 KB
  __shared__ unsigned short ldsB[2][BN * BK];   // 2 x  8 KB
  int lane = threadIdx.x & 31;
  int wave = threadIdx.x >> 5;
  int g = lane >> 4;                  // lane group
  int r = lane & 15;
  int wr = wave >> 1;                 // 0..3 -> M sub-tile pair
  int wc = wave & 1;                  // 0..1 -> N sub-tile pair
  int m0 = blockIdx.y * BM;
  int n0 = blockIdx.x * BN;
  if (m0 >= M || n0 >= N) return;

  unsigned aA[2] = { (unsigned)(size_t)(void*)ldsA[0], (unsigned)(size_t)(void*)ldsA[1] };
  unsigned aB[2] = { (unsigned)(size_t)(void*)ldsB[0], (unsigned)(size_t)(void*)ldsB[1] };
  const unsigned short* Abase = A  + (size_t)m0 * K;
  const unsigned short* Bbase = Bt + (size_t)n0 * K;
  int nk = K / BK;

  // prologue: fetch chunk 0 into buffer 0
  if (threadIdx.x < 32) {
    tdm_load_2d(aA[0], Abase, BK, BM, (unsigned)K, (unsigned)K, 1u << 20);
    tdm_load_2d(aB[0], Bbase, BK, BN, (unsigned)K, (unsigned)K, 1u << 20);
    __builtin_amdgcn_s_wait_tensorcnt(0);
  }
  __syncthreads();                    // publish buffer 0

  v8f acc[2][2] = {};
  for (int t = 0; t < nk; ++t) {
    int cur = t & 1, nxt = cur ^ 1;
    // kick off DMA for chunk t+1 while computing on chunk t
    if (t + 1 < nk && threadIdx.x < 32) {
      tdm_load_2d(aA[nxt], Abase + (size_t)(t + 1) * BK, BK, BM,
                  (unsigned)K, (unsigned)K, 1u << 20);
      tdm_load_2d(aB[nxt], Bbase + (size_t)(t + 1) * BK, BK, BN,
                  (unsigned)K, (unsigned)K, 1u << 20);
    }
    const unsigned short* tA = ldsA[cur];
    const unsigned short* tB = ldsB[cur];
#pragma unroll
    for (int kk = 0; kk < BK; kk += 32) {
      FragBF fa[2], fb[2];
#pragma unroll
      for (int i = 0; i < 2; ++i) {
        // A 16x32 frag: elems 0..7 -> K=kk+8g+e ; 8..15 -> K=kk+16+8g+(e-8)
        const unsigned short* ar = &tA[((wr * 2 + i) * 16 + r) * BK + kk];
        fa[i].u[0] = *(const v4u*)(ar + 8 * g);
        fa[i].u[1] = *(const v4u*)(ar + 16 + 8 * g);
        // B 32x16 frag from N-major rows: elems 0..7 -> K=kk+8g+e ; 8..15 -> K=kk+16+8g+(e-8)
        const unsigned short* br = &tB[((wc * 2 + i) * 16 + r) * BK + kk];
        fb[i].u[0] = *(const v4u*)(br + 8 * g);
        fb[i].u[1] = *(const v4u*)(br + 16 + 8 * g);
      }
#pragma unroll
      for (int i = 0; i < 2; ++i)
#pragma unroll
        for (int j = 0; j < 2; ++j)
          acc[i][j] = __builtin_amdgcn_wmma_f32_16x16x32_bf16(
              false, fa[i].bf, false, fb[j].bf, (short)0, acc[i][j], false, false);
    }
    __syncthreads();                  // all waves done with buffer cur
    if (t + 1 < nk) {
      if (threadIdx.x < 32) __builtin_amdgcn_s_wait_tensorcnt(0);
      __syncthreads();                // publish buffer nxt
    }
  }
#pragma unroll
  for (int i = 0; i < 2; ++i) {
    int row0 = m0 + (wr * 2 + i) * 16 + 8 * g;   // C/D: VGPR q -> row0+q, col
#pragma unroll
    for (int j = 0; j < 2; ++j) {
      int col = n0 + (wc * 2 + j) * 16 + r;
      float* cp = C + (size_t)row0 * N + col;
#pragma unroll
      for (int q = 0; q < 8; ++q) cp[(size_t)q * N] = acc[i][j][q];
    }
  }
}

// ---------------- GAT attention pieces ----------------
__global__ void attn_scores(const float* __restrict__ H, const float* __restrict__ a_src,
                            const float* __restrict__ a_dst, float* __restrict__ es,
                            float* __restrict__ ed, int heads, int outc) {
  int wid = (int)(((size_t)blockIdx.x * blockDim.x + threadIdx.x) >> 5);
  int lane = threadIdx.x & 31;
  if (wid >= N_NODES * heads) return;
  int node = wid / heads, h = wid % heads;
  const float* hp = H + (size_t)node * heads * outc + (size_t)h * outc;
  const float* as = a_src + h * outc;
  const float* ad = a_dst + h * outc;
  float s = 0.f, d = 0.f;
  for (int c = lane; c < outc; c += 32) { float v = hp[c]; s += v * as[c]; d += v * ad[c]; }
  for (int off = 16; off > 0; off >>= 1) { s += __shfl_down(s, off, 32); d += __shfl_down(d, off, 32); }
  if (lane == 0) { es[node * heads + h] = s; ed[node * heads + h] = d; }
}

__global__ void edge_max(const float* __restrict__ es, const float* __restrict__ ed,
                         const int* __restrict__ src, const int* __restrict__ dst,
                         float* __restrict__ m, int heads) {
  int idx = blockIdx.x * blockDim.x + threadIdx.x;
  if (idx >= ET * heads) return;
  int e = idx / heads, h = idx % heads;
  int s = e < NE ? src[e] : (e - NE);
  int d = e < NE ? dst[e] : (e - NE);
  float ev = es[s * heads + h] + ed[d * heads + h];
  ev = ev > 0.f ? ev : 0.2f * ev;          // leaky_relu 0.2
  atomicMaxF(&m[d * heads + h], ev);
}

__global__ void edge_prob(const float* __restrict__ es, const float* __restrict__ ed,
                          const int* __restrict__ src, const int* __restrict__ dst,
                          const float* __restrict__ m, float* __restrict__ z,
                          float* __restrict__ p, int heads) {
  int idx = blockIdx.x * blockDim.x + threadIdx.x;
  if (idx >= ET * heads) return;
  int e = idx / heads, h = idx % heads;
  int s = e < NE ? src[e] : (e - NE);
  int d = e < NE ? dst[e] : (e - NE);
  float ev = es[s * heads + h] + ed[d * heads + h];
  ev = ev > 0.f ? ev : 0.2f * ev;
  float pv = __expf(ev - m[d * heads + h]);
  p[idx] = pv;
  atomicAdd(&z[d * heads + h], pv);
}

__global__ void edge_aggregate(const float* __restrict__ p, const float* __restrict__ z,
                               const float* __restrict__ H, const int* __restrict__ src,
                               const int* __restrict__ dst, float* __restrict__ out,
                               int heads, int outc) {
  int wid = (int)(((size_t)blockIdx.x * blockDim.x + threadIdx.x) >> 5);
  int lane = threadIdx.x & 31;
  if (wid >= ET * heads) return;
  int e = wid / heads, h = wid % heads;
  int s = e < NE ? src[e] : (e - NE);
  int d = e < NE ? dst[e] : (e - NE);
  float alpha = p[wid] / z[d * heads + h];
  const float* hs = H + (size_t)s * heads * outc + (size_t)h * outc;
  float* od = out + (size_t)d * heads * outc + (size_t)h * outc;
  for (int c = lane; c < outc; c += 32) atomicAdd(&od[c], alpha * hs[c]);
}

__global__ void bias_relu_store(const float* __restrict__ acc, const float* __restrict__ bias,
                                int F, float* __restrict__ out_f32,
                                unsigned short* __restrict__ out_bf, size_t total) {
  size_t i = (size_t)blockIdx.x * blockDim.x + threadIdx.x;
  if (i >= total) return;
  int f = (int)(i % F);
  float v = acc[i] + bias[f];
  v = v > 0.f ? v : 0.f;
  if (out_f32) out_f32[i] = v;
  if (out_bf)  out_bf[i]  = f2bf(v);
}

// ---------------- classifier softmax S (K, N, 2) ----------------
__global__ void clf_softmax(const float* __restrict__ x, const float* __restrict__ w,
                            const float* __restrict__ bcls, float* __restrict__ S) {
  int wid = (int)(((size_t)blockIdx.x * blockDim.x + threadIdx.x) >> 5);
  int lane = threadIdx.x & 31;
  if (wid >= KCH * N_NODES) return;
  int k = wid / N_NODES, n = wid % N_NODES;
  const float* xr = x + (size_t)n * HID;
  const float* wk = w + (size_t)k * HID * 2;
  float l0 = 0.f, l1 = 0.f;
  for (int h = lane; h < HID; h += 32) { float xv = xr[h]; l0 += xv * wk[h * 2]; l1 += xv * wk[h * 2 + 1]; }
  for (int off = 16; off > 0; off >>= 1) { l0 += __shfl_down(l0, off, 32); l1 += __shfl_down(l1, off, 32); }
  if (lane == 0) {
    l0 += bcls[k * 2]; l1 += bcls[k * 2 + 1];
    float mx = fmaxf(l0, l1);
    float e0 = __expf(l0 - mx), e1 = __expf(l1 - mx);
    float inv = 1.f / (e0 + e1);
    size_t base = ((size_t)k * N_NODES + n) * 2;
    S[base] = e0 * inv; S[base + 1] = e1 * inv;
  }
}

// ---------------- adjacency aggregation / loss ----------------
__global__ void edge_adj(const float* __restrict__ S, const int* __restrict__ src,
                         const int* __restrict__ dst, const int* __restrict__ batch,
                         float* __restrict__ agg) {
  int e = blockIdx.x * blockDim.x + threadIdx.x;
  if (e >= NE) return;
  int s = src[e], d = dst[e];
  int bs = batch[s];
  if (bs != batch[d]) return;
  for (int k = 0; k < KCH; ++k) {
    const float* Ss = S + ((size_t)k * N_NODES + s) * 2;
    const float* Sd = S + ((size_t)k * N_NODES + d) * 2;
    float* a = agg + ((size_t)bs * KCH + k) * 4;
    atomicAdd(&a[0], Ss[0] * Sd[0]);
    atomicAdd(&a[1], Ss[0] * Sd[1]);
    atomicAdd(&a[2], Ss[1] * Sd[0]);
    atomicAdd(&a[3], Ss[1] * Sd[1]);
  }
}

__global__ void agg_loss_kernel(const float* __restrict__ agg, float* __restrict__ agg_loss) {
  int idx = blockIdx.x * blockDim.x + threadIdx.x;
  if (idx >= NB * KCH) return;
  int k = idx % KCH;
  const float* a = agg + (size_t)idx * 4;
  float part = 0.f;
  for (int i = 0; i < 2; ++i) {
    float denom = fmaxf(fabsf(a[i * 2]) + fabsf(a[i * 2 + 1]), 1e-5f);
    float diag = a[i * 2 + i] / denom;
    part += (diag - 1.f) * (diag - 1.f);
  }
  atomicAdd(&agg_loss[k], part / (float)(NB * 2));
}

// ---------------- pooling ----------------
__global__ void pool_g(const float* __restrict__ x, float* __restrict__ g_emb) {
  int b = blockIdx.x, c = threadIdx.x;
  float s = 0.f;
  const float* base = x + (size_t)b * SZ * HID + c;
  for (int i = 0; i < SZ; ++i) s += base[(size_t)i * HID];
  g_emb[b * HID + c] = s / (float)SZ;
}

__global__ void pool_s(const float* __restrict__ x, const float* __restrict__ S,
                       float* __restrict__ s_emb) {
  int bk = blockIdx.x; int b = bk / KCH, k = bk % KCH;
  int c = threadIdx.x;
  float s = 0.f;
  for (int i = 0; i < SZ; ++i) {
    int n = b * SZ + i;
    s += S[((size_t)k * N_NODES + n) * 2] * x[(size_t)n * HID + c];
  }
  s_emb[((size_t)b * KCH + k) * HID + c] = s;
}

// ---------------- MI estimator ----------------
__global__ void estimator(const float* __restrict__ g_emb, const float* __restrict__ s_emb,
                          const float* __restrict__ ib1_w, const float* __restrict__ ib1_b,
                          const float* __restrict__ ib2_w, const float* __restrict__ ib2_b,
                          const float* __restrict__ bn1_w, const float* __restrict__ bn1_b,
                          const float* __restrict__ bn1_m, const float* __restrict__ bn1_v,
                          const float* __restrict__ bn2_w, const float* __restrict__ bn2_b,
                          const float* __restrict__ bn2_m, const float* __restrict__ bn2_v,
                          float* __restrict__ je, float* __restrict__ me) {
  int blk = blockIdx.x;
  int which = blk / (KCH * NB);
  int rem = blk % (KCH * NB);
  int k = rem / NB, b = rem % NB;
  int t = threadIdx.x;
  __shared__ float zv[2 * HID];
  __shared__ float red[HID];
  int gb = (which == 0) ? b : ((b * 7 + k * 13 + k) % NB);  // deterministic perm stand-in
  zv[t]       = g_emb[gb * HID + t];
  zv[t + HID] = s_emb[((size_t)b * KCH + k) * HID + t];
  __syncthreads();
  float acc = ib1_b[t];
  for (int j = 0; j < 2 * HID; ++j) acc += zv[j] * ib1_w[(size_t)j * HID + t];
  acc = acc > 0.f ? acc : 0.f;
  acc = (acc - bn1_m[t]) * rsqrtf(bn1_v[t] + 1e-5f) * bn1_w[t] + bn1_b[t];
  red[t] = acc * ib2_w[t];
  __syncthreads();
  for (int off = 128; off > 0; off >>= 1) {
    if (t < off) red[t] += red[t + off];
    __syncthreads();
  }
  if (t == 0) {
    float h = red[0] + ib2_b[0];
    h = h > 0.f ? h : 0.f;
    h = (h - bn2_m[0]) * rsqrtf(bn2_v[0] + 1e-5f) * bn2_w[0] + bn2_b[0];
    (which == 0 ? je : me)[k * NB + b] = h;
  }
}

__global__ void mi_kernel(const float* __restrict__ je, const float* __restrict__ me,
                          float* __restrict__ mi) {
  int k = blockIdx.x, t = threadIdx.x;
  __shared__ float r1[256], r2[256];
  float se = 0.f, sj = 0.f;
  if (t < NB) { se = __expf(me[k * NB + t]); sj = je[k * NB + t]; }
  r1[t] = se; r2[t] = sj;
  __syncthreads();
  for (int off = 128; off > 0; off >>= 1) {
    if (t < off) { r1[t] += r1[t + off]; r2[t] += r2[t + off]; }
    __syncthreads();
  }
  if (t == 0) {
    float mexp = fminf(fmaxf(r1[0] / (float)NB, 1.0f), 1e25f);
    float v = fminf(fmaxf(logf(mexp), -100000.f), 100000.f);
    mi[k] = v - r2[0] / (float)NB;
  }
}

// ---------------- diversity ----------------
__global__ void norms_kernel(const float* __restrict__ s_emb, float* __restrict__ norms) {
  int idx = blockIdx.x * blockDim.x + threadIdx.x;
  if (idx >= NB * KCH) return;
  const float* v = s_emb + (size_t)idx * HID;
  float s = 0.f;
  for (int c = 0; c < HID; ++c) s += v[c] * v[c];
  norms[idx] = fmaxf(sqrtf(s), 1e-8f);
}

__global__ void div_kernel(const float* __restrict__ s_emb, const float* __restrict__ norms,
                           float* __restrict__ divacc) {
  int c = blockIdx.x;
  int k = 0;
  while (c >= KCH - 1 - k) { c -= KCH - 1 - k; ++k; }
  int l = k + 1 + c;
  int t = threadIdx.x;
  __shared__ float red[256];
  float s = 0.f;
  if (t < NB) {
    const float* a  = s_emb + ((size_t)t * KCH + k) * HID;
    const float* bb = s_emb + ((size_t)t * KCH + l) * HID;
    float d = 0.f;
    for (int cc = 0; cc < HID; ++cc) d += a[cc] * bb[cc];
    s = d / (norms[t * KCH + k] * norms[t * KCH + l]);
  }
  red[t] = s;
  __syncthreads();
  for (int off = 128; off > 0; off >>= 1) { if (t < off) red[t] += red[t + off]; __syncthreads(); }
  if (t == 0) atomicAdd(divacc, red[0] / (float)(NB * (KCH * (KCH - 1) / 2)));
}

// ---------------- selection + head ----------------
__global__ void argmin_kernel(const float* __restrict__ mi, const float* __restrict__ agg_loss,
                              const float* __restrict__ divacc, int* __restrict__ idxbuf,
                              float* __restrict__ out_loss) {
  float best = mi[0]; int bi = 0;
  for (int k = 1; k < KCH; ++k) { float v = mi[k]; if (v < best) { best = v; bi = k; } }
  *idxbuf = bi;
  *out_loss = agg_loss[bi] + *divacc + mi[bi];
}

__global__ void head_out(const float* __restrict__ s_emb, const int* __restrict__ idxbuf,
                         const float* __restrict__ fc1_w, const float* __restrict__ fc1_b,
                         const float* __restrict__ fc2_w, const float* __restrict__ fc2_b,
                         float* __restrict__ out) {
  int b = blockIdx.x, t = threadIdx.x;
  __shared__ float sel[HID];
  __shared__ float r0[HID], r1[HID];
  int idx = *idxbuf;
  sel[t] = s_emb[((size_t)b * KCH + idx) * HID + t];
  __syncthreads();
  float acc = fc1_b[t];
  for (int j = 0; j < HID; ++j) acc += sel[j] * fc1_w[(size_t)j * HID + t];
  acc = acc > 0.f ? acc : 0.f;
  r0[t] = acc * fc2_w[t * 2];
  r1[t] = acc * fc2_w[t * 2 + 1];
  __syncthreads();
  for (int off = 128; off > 0; off >>= 1) {
    if (t < off) { r0[t] += r0[t + off]; r1[t] += r1[t + off]; }
    __syncthreads();
  }
  if (t == 0) { out[b * 2] = r0[0] + fc2_b[0]; out[b * 2 + 1] = r1[0] + fc2_b[1]; }
}

// ---------------- host orchestration ----------------
extern "C" void kernel_launch(void* const* d_in, const int* in_sizes, int n_in,
                              void* d_out, int out_size, void* d_ws, size_t ws_size,
                              hipStream_t stream) {
  (void)in_sizes; (void)n_in; (void)out_size; (void)ws_size;
  const float* node_x   = (const float*)d_in[0];
  const int*   edge_idx = (const int*)d_in[1];
  const int*   src      = edge_idx;
  const int*   dst      = edge_idx + NE;
  const int*   batch    = (const int*)d_in[2];
  const float* W1       = (const float*)d_in[3];
  const float* att_src1 = (const float*)d_in[4];
  const float* att_dst1 = (const float*)d_in[5];
  const float* b1       = (const float*)d_in[6];
  const float* W2       = (const float*)d_in[7];
  const float* att_src2 = (const float*)d_in[8];
  const float* att_dst2 = (const float*)d_in[9];
  const float* b2       = (const float*)d_in[10];
  const float* clf_w    = (const float*)d_in[11];
  const float* clf_b    = (const float*)d_in[12];
  const float* fc1_w    = (const float*)d_in[13];
  const float* fc1_b    = (const float*)d_in[14];
  const float* fc2_w    = (const float*)d_in[15];
  const float* fc2_b    = (const float*)d_in[16];
  const float* ib1_w    = (const float*)d_in[17];
  const float* ib1_b    = (const float*)d_in[18];
  const float* ib2_w    = (const float*)d_in[19];
  const float* ib2_b    = (const float*)d_in[20];
  const float* bn1_w    = (const float*)d_in[21];
  const float* bn1_b    = (const float*)d_in[22];
  const float* bn1_m    = (const float*)d_in[23];
  const float* bn1_v    = (const float*)d_in[24];
  const float* bn2_w    = (const float*)d_in[25];
  const float* bn2_b    = (const float*)d_in[26];
  const float* bn2_m    = (const float*)d_in[27];
  const float* bn2_v    = (const float*)d_in[28];
  float* out = (float*)d_out;

  // -------- workspace carve --------
  char* ws = (char*)d_ws;
  size_t off = 0;
  auto alloc = [&](size_t bytes) -> void* {
    void* p = (void*)(ws + off);
    off += (bytes + 255) & ~(size_t)255;
    return p;
  };
  const int F1 = HEADS * HID;                         // 1024
  unsigned short* x1bf = (unsigned short*)alloc((size_t)MP * F_IN * 2);       // padded rows
  unsigned short* w1t  = (unsigned short*)alloc((size_t)F1 * F_IN * 2);
  float* h1   = (float*)alloc((size_t)MP * F1 * 4);                           // padded rows
  float* es1  = (float*)alloc((size_t)N_NODES * HEADS * 4);
  float* ed1  = (float*)alloc((size_t)N_NODES * HEADS * 4);
  float* m1   = (float*)alloc((size_t)N_NODES * HEADS * 4);
  float* z1   = (float*)alloc((size_t)N_NODES * HEADS * 4);
  float* p1   = (float*)alloc((size_t)ET * HEADS * 4);
  float* acc1 = (float*)alloc((size_t)N_NODES * F1 * 4);
  unsigned short* x2bf = (unsigned short*)alloc((size_t)MP * F1 * 2);         // padded rows
  unsigned short* w2t  = (unsigned short*)alloc((size_t)HID * F1 * 2);
  float* h2   = (float*)alloc((size_t)MP * HID * 4);                          // padded rows
  float* es2  = (float*)alloc((size_t)N_NODES * 4);
  float* ed2  = (float*)alloc((size_t)N_NODES * 4);
  float* m2   = (float*)alloc((size_t)N_NODES * 4);
  float* z2   = (float*)alloc((size_t)N_NODES * 4);
  float* p2   = (float*)alloc((size_t)ET * 4);
  float* x3   = (float*)alloc((size_t)N_NODES * HID * 4);   // acc2 -> in-place relu
  float* S    = (float*)alloc((size_t)KCH * N_NODES * 2 * 4);
  float* agg  = (float*)alloc((size_t)NB * KCH * 4 * 4);
  float* aggL = (float*)alloc((size_t)KCH * 4);
  float* gemb = (float*)alloc((size_t)NB * HID * 4);
  float* semb = (float*)alloc((size_t)NB * KCH * HID * 4);
  float* nrm  = (float*)alloc((size_t)NB * KCH * 4);
  float* je   = (float*)alloc((size_t)KCH * NB * 4);
  float* me   = (float*)alloc((size_t)KCH * NB * 4);
  float* mi   = (float*)alloc((size_t)KCH * 4);
  float* dv   = (float*)alloc(4);
  int*   idxb = (int*)alloc(4);

  const int T = 256;
  auto nb = [](size_t n, int t) { return (unsigned)((n + (size_t)t - 1) / t); };

  // -------- layer 1 --------
  cvt_bf16<<<nb((size_t)N_NODES * F_IN, T), T, 0, stream>>>(node_x, x1bf, (size_t)N_NODES * F_IN);
  cvt_bf16_transpose<<<nb((size_t)F_IN * F1, T), T, 0, stream>>>(W1, w1t, F_IN, F1);
  {
    dim3 grid(F1 / BN, MP / BM);
    gemm_bf16_wmma<<<grid, 256, 0, stream>>>(x1bf, w1t, h1, MP, F_IN, F1);
  }
  attn_scores<<<nb((size_t)N_NODES * HEADS * 32, T), T, 0, stream>>>(h1, att_src1, att_dst1, es1, ed1, HEADS, HID);
  fill_f32<<<nb((size_t)N_NODES * HEADS, T), T, 0, stream>>>(m1, -3.0e38f, (size_t)N_NODES * HEADS);
  fill_f32<<<nb((size_t)N_NODES * HEADS, T), T, 0, stream>>>(z1, 0.f, (size_t)N_NODES * HEADS);
  fill_f32<<<nb((size_t)N_NODES * F1, T), T, 0, stream>>>(acc1, 0.f, (size_t)N_NODES * F1);
  edge_max<<<nb((size_t)ET * HEADS, T), T, 0, stream>>>(es1, ed1, src, dst, m1, HEADS);
  edge_prob<<<nb((size_t)ET * HEADS, T), T, 0, stream>>>(es1, ed1, src, dst, m1, z1, p1, HEADS);
  edge_aggregate<<<nb((size_t)ET * HEADS * 32, T), T, 0, stream>>>(p1, z1, h1, src, dst, acc1, HEADS, HID);
  bias_relu_store<<<nb((size_t)N_NODES * F1, T), T, 0, stream>>>(acc1, b1, F1, nullptr, x2bf, (size_t)N_NODES * F1);

  // -------- layer 2 --------
  cvt_bf16_transpose<<<nb((size_t)F1 * HID, T), T, 0, stream>>>(W2, w2t, F1, HID);
  {
    dim3 grid(HID / BN, MP / BM);
    gemm_bf16_wmma<<<grid, 256, 0, stream>>>(x2bf, w2t, h2, MP, F1, HID);
  }
  attn_scores<<<nb((size_t)N_NODES * 32, T), T, 0, stream>>>(h2, att_src2, att_dst2, es2, ed2, 1, HID);
  fill_f32<<<nb((size_t)N_NODES, T), T, 0, stream>>>(m2, -3.0e38f, (size_t)N_NODES);
  fill_f32<<<nb((size_t)N_NODES, T), T, 0, stream>>>(z2, 0.f, (size_t)N_NODES);
  fill_f32<<<nb((size_t)N_NODES * HID, T), T, 0, stream>>>(x3, 0.f, (size_t)N_NODES * HID);
  edge_max<<<nb((size_t)ET, T), T, 0, stream>>>(es2, ed2, src, dst, m2, 1);
  edge_prob<<<nb((size_t)ET, T), T, 0, stream>>>(es2, ed2, src, dst, m2, z2, p2, 1);
  edge_aggregate<<<nb((size_t)ET * 32, T), T, 0, stream>>>(p2, z2, h2, src, dst, x3, 1, HID);
  bias_relu_store<<<nb((size_t)N_NODES * HID, T), T, 0, stream>>>(x3, b2, HID, x3, nullptr, (size_t)N_NODES * HID);

  // -------- classifier / adjacency --------
  clf_softmax<<<nb((size_t)KCH * N_NODES * 32, T), T, 0, stream>>>(x3, clf_w, clf_b, S);
  fill_f32<<<nb((size_t)NB * KCH * 4, T), T, 0, stream>>>(agg, 0.f, (size_t)NB * KCH * 4);
  fill_f32<<<1, T, 0, stream>>>(aggL, 0.f, (size_t)KCH);
  fill_f32<<<1, T, 0, stream>>>(dv, 0.f, (size_t)1);
  edge_adj<<<nb((size_t)NE, T), T, 0, stream>>>(S, src, dst, batch, agg);
  agg_loss_kernel<<<nb((size_t)NB * KCH, T), T, 0, stream>>>(agg, aggL);

  // -------- pooling / MI / diversity --------
  pool_g<<<NB, HID, 0, stream>>>(x3, gemb);
  pool_s<<<NB * KCH, HID, 0, stream>>>(x3, S, semb);
  estimator<<<2 * KCH * NB, 256, 0, stream>>>(gemb, semb, ib1_w, ib1_b, ib2_w, ib2_b,
                                              bn1_w, bn1_b, bn1_m, bn1_v,
                                              bn2_w, bn2_b, bn2_m, bn2_v, je, me);
  mi_kernel<<<KCH, 256, 0, stream>>>(je, me, mi);
  norms_kernel<<<nb((size_t)NB * KCH, T), T, 0, stream>>>(semb, nrm);
  div_kernel<<<KCH * (KCH - 1) / 2, 256, 0, stream>>>(semb, nrm, dv);

  // -------- select + head --------
  argmin_kernel<<<1, 1, 0, stream>>>(mi, aggL, dv, idxb, out + NB * 2);
  head_out<<<NB, HID, 0, stream>>>(semb, idxb, fc1_w, fc1_b, fc2_w, fc2_b, out);
}
